// GetInflPredictions3d_53635551592645
// MI455X (gfx1250) — matrix operand order repaired and checked by reading, not compiled, not run on hardware
//
#include <hip/hip_runtime.h>
#include <hip/hip_bf16.h>
#include <math.h>

typedef __attribute__((ext_vector_type(16))) _Float16 v16h;
typedef __attribute__((ext_vector_type(8)))  float    v8f;

#define C81  81
#define NPAD 96   // 81 padded to 6 tiles of 16

// ---------------------------------------------------------------------------
// Kernel 1: per-center geometry. Block c computes the stable dist-rank order,
// sorted base weights and sorted angles for center c. Exact integer d^2 is
// used for ranking (sqrt is monotone & injective on these values), index
// tie-break reproduces jnp.argsort(stable). Also zeroes d_out.
// ---------------------------------------------------------------------------
__global__ __launch_bounds__(96) void setup_kernel(const float* __restrict__ board,
                                                   int*   __restrict__ order,
                                                   float* __restrict__ base,
                                                   float* __restrict__ angle,
                                                   float* __restrict__ out) {
    const int c = blockIdx.x;
    const int j = threadIdx.x;
    __shared__ int   d2s[NPAD];
    __shared__ float angs[NPAD];

    if (c == 0 && j < C81) out[j] = 0.0f;

    const int cy = c / 9, cx = c % 9;
    if (j < C81) {
        const int jy = j / 9, jx = j % 9;
        const int dy = jy - cy, dx = jx - cx;
        d2s[j] = dy * dy + dx * dx;
        // mirror reference f32 math exactly (incl. -0.0 behavior of -dy)
        float fdy = (float)dy, fdx = (float)dx;
        float raw = atan2f(-fdy, fdx) * 57.29577951308232f;   // 180/pi as f32
        angs[j] = (raw > 0.0f) ? raw : raw + 360.0f;
    }
    __syncthreads();

    if (j < C81) {
        const int myd = d2s[j];
        int rank = 0;
        #pragma unroll 1
        for (int i = 0; i < C81; ++i) {
            const int di = d2s[i];
            rank += (di < myd) || (di == myd && i < j);
        }
        const float MAXD = 12.727922061357855f;               // sqrt(2)*9 as f32
        float dist = sqrtf((float)myd);
        float bse  = (MAXD - dist) / MAXD;
        if (bse < 0.5f) bse *= 0.5f;                          // DIST_LIN_W
        order[c * NPAD + rank] = j;
        base [c * NPAD + rank] = bse;
        angle[c * NPAD + rank] = angs[j];
    }
}

// element -> K mapping for the 16-bit WMMA A-operand (ISA 7.12.2)
__device__ __forceinline__ int kmapA(int e, int half) {
    int v = e >> 1;
    return ((v < 4) ? (2 * v) : (8 + 2 * v)) + (e & 1) + 8 * half;
}

// ---------------------------------------------------------------------------
// Kernel 2: block c (skipped unless board[c]==0) computes, via f16 WMMA,
//   K1 = P_c * A_c ,  K2 = N_c * A_c   (96x96x96 each, exact small integers)
// then infl[b,j] = base_c[j] * 2^-(n_j*K1 + p_j*K2) * (p_j - n_j), summed over
// j and atomically accumulated into out[b] for empty b != c.
// ---------------------------------------------------------------------------
__global__ __launch_bounds__(256) void influence_kernel(const float* __restrict__ board,
                                                        const int*   __restrict__ order,
                                                        const float* __restrict__ base,
                                                        const float* __restrict__ angle,
                                                        float* __restrict__ out) {
    const int c = blockIdx.x;
    if (board[c] != 0.0f) return;   // block-uniform: only empty centers contribute

    __shared__ _Float16 Ac[NPAD * NPAD];   // A_c[i][j]
    __shared__ _Float16 Pm[NPAD * NPAD];   // p[b][i]
    __shared__ _Float16 Nm[NPAD * NPAD];   // n[b][i]
    __shared__ float rowsum[NPAD];
    __shared__ float baseRow[NPAD];
    __shared__ float angRow[NPAD];
    __shared__ int   ordRow[NPAD];
    __shared__ float bfs[NPAD];

    const int tid = threadIdx.x;

    for (int q = tid; q < NPAD; q += blockDim.x) {
        rowsum[q]  = 0.0f;
        bfs[q]     = (q < C81) ? board[q] : 0.0f;
        baseRow[q] = (q < C81) ? base [c * NPAD + q] : 0.0f;
        angRow[q]  = (q < C81) ? angle[c * NPAD + q] : 0.0f;
        ordRow[q]  = (q < C81) ? order[c * NPAD + q] : 0;
    }
    __syncthreads();

    for (int idx = tid; idx < NPAD * NPAD; idx += blockDim.x) {
        const int r0 = idx / NPAD;          // A: i   ; P/N: b
        const int c0 = idx % NPAD;          // A: j   ; P/N: i (sorted pos)
        _Float16 a = (_Float16)0.0f;
        if (r0 < C81 && c0 < C81 && r0 < c0) {
            float ad = fabsf(angRow[r0] - angRow[c0]);
            if (ad > 180.0f) ad = 360.0f - ad;
            if (ad < 45.0f) a = (_Float16)1.0f;          // ANGLE_LT_W
        }
        Ac[idx] = a;

        _Float16 p = (_Float16)0.0f, n = (_Float16)0.0f;
        if (r0 < C81 && c0 < C81) {
            const int   o = ordRow[c0];
            const float v = bfs[o];
            if (o == r0) { p = (_Float16)fmaxf(v + 1.0f, 0.0f); }  // STONE_VALUE
            else         { p = (_Float16)fmaxf(v, 0.0f);
                           n = (_Float16)fmaxf(-v, 0.0f); }
        }
        Pm[idx] = p;
        Nm[idx] = n;
    }
    __syncthreads();

    const int wave = tid >> 5;
    const int lane = tid & 31;
    const int half = lane >> 4;
    const int lr   = lane & 15;

    // 36 output tiles of 16x16, 8 waves; loop bounds are wave-uniform so
    // EXEC stays all-ones across every v_wmma (ISA requirement).
    for (int t = wave; t < 36; t += 8) {
        const int tm = t / 6, tn = t % 6;
        v8f acc1 = {};   // (P * A)   tile
        v8f acc2 = {};   // (N * A)   tile
        #pragma unroll
        for (int ks = 0; ks < 3; ++ks) {
            const int ko = ks * 32;
            v16h aP, aN, bA;
            const int mrow = tm * 16 + lr;
            #pragma unroll
            for (int e = 0; e < 16; ++e) {
                const int K = kmapA(e, half);
                aP[e] = Pm[mrow * NPAD + ko + K];
                aN[e] = Nm[mrow * NPAD + ko + K];
            }
            const int ncol = tn * 16 + lr;   // B: lane = N, element = K (same map as A)
            #pragma unroll
            for (int e = 0; e < 16; ++e)
                bA[e] = Ac[(ko + kmapA(e, half)) * NPAD + ncol];

            acc1 = __builtin_amdgcn_wmma_f32_16x16x32_f16(false, aP, false, bA,
                                                          (short)0, acc1, false, false);
            acc2 = __builtin_amdgcn_wmma_f32_16x16x32_f16(false, aN, false, bA,
                                                          (short)0, acc2, false, false);
        }
        // Epilogue: D layout — VGPR r, lane l -> (M = r + 8*(l>>4), N = l&15)
        const int j = tn * 16 + lr;
        if (j < C81) {
            const float bj = baseRow[j];
            #pragma unroll
            for (int r = 0; r < 8; ++r) {
                const int b = tm * 16 + r + 8 * half;
                if (b < C81 && b != c && bfs[b] == 0.0f) {
                    const float pj = (float)Pm[b * NPAD + j];
                    const float nj = (float)Nm[b * NPAD + j];
                    const float s  = pj - nj;
                    if (s != 0.0f) {
                        const float k    = nj * acc1[r] + pj * acc2[r]; // exact int
                        const float infl = bj * exp2f(-k) * s;          // 0.5^k
                        atomicAdd(&rowsum[b], infl);
                    }
                }
            }
        }
    }
    __syncthreads();

    for (int b = tid; b < C81; b += blockDim.x)
        if (rowsum[b] != 0.0f) atomicAdd(&out[b], rowsum[b]);
}

extern "C" void kernel_launch(void* const* d_in, const int* in_sizes, int n_in,
                              void* d_out, int out_size, void* d_ws, size_t ws_size,
                              hipStream_t stream) {
    (void)in_sizes; (void)n_in; (void)out_size; (void)ws_size;
    const float* board = (const float*)d_in[1];   // d_in[0] = all_coords (unused: derivable)
    int*   order = (int*)d_ws;
    float* base  = (float*)((char*)d_ws + (size_t)NPAD * NPAD * sizeof(int));
    float* angle = base + NPAD * NPAD;
    float* out   = (float*)d_out;

    setup_kernel    <<<C81,  96, 0, stream>>>(board, order, base, angle, out);
    influence_kernel<<<C81, 256, 0, stream>>>(board, order, base, angle, out);
}
